// transformerModel_74955769250658
// MI455X (gfx1250) — compile-verified, matrix operand order checked by reading, not run on hardware
//
#include <hip/hip_runtime.h>

#define L_  6
#define B_  4
#define S_  1024
#define D_  1024
#define H_  16
#define HD_ 64
#define F_  4096
#define V_  32000
#define M_  (B_*S_)   // 4096 rows

typedef _Float16 v16h __attribute__((ext_vector_type(16)));
typedef _Float16 v8h  __attribute__((ext_vector_type(8)));
typedef float    v8f  __attribute__((ext_vector_type(8)));
typedef int      v4i  __attribute__((vector_size(16)));

union Frag { v16h v; v8h h[2]; };

__device__ __forceinline__ v8f wmma_f16(const Frag& a, const Frag& b, v8f c) {
  // D = A(16x32 f16) x B(32x16 f16) + C(16x16 f32)
  return __builtin_amdgcn_wmma_f32_16x16x32_f16(false, a.v, false, b.v,
                                                (short)0, c, false, false);
}

// --- CDNA5 async global->LDS copy path (ASYNCcnt-tracked), with fallback ----
#if defined(__gfx1250__) && __has_builtin(__builtin_amdgcn_global_load_async_to_lds_b128)
#define ASYNC_LDS 1
#else
#define ASYNC_LDS 0
#endif

__device__ __forceinline__ void async_cp16(const _Float16* g, _Float16* l) {
#if ASYNC_LDS
  __builtin_amdgcn_global_load_async_to_lds_b128(
      (__attribute__((address_space(1))) v4i*)g,
      (__attribute__((address_space(3))) v4i*)l, 0, 0);
#else
  (void)g; (void)l;
#endif
}

__device__ __forceinline__ void async_wait0() {
#if defined(__gfx1250__) && __has_builtin(__builtin_amdgcn_s_wait_asynccnt)
  __builtin_amdgcn_s_wait_asynccnt(0);
#else
  asm volatile("s_wait_asynccnt 0x0" ::: "memory");
#endif
}

// ---------------------------------------------------------------------------
// x = tok_emb[tokens] + pos_emb ; also f16 copy
// ---------------------------------------------------------------------------
__global__ __launch_bounds__(256) void embed_kernel(
    const int* __restrict__ tokens, const float* __restrict__ te,
    const float* __restrict__ pe, float* __restrict__ x,
    _Float16* __restrict__ xh)
{
  size_t idx = (size_t)blockIdx.x * 256 + threadIdx.x;   // over M_*D_
  int m = (int)(idx >> 10);
  int d = (int)(idx & 1023);
  int s = m & (S_ - 1);
  int tok = tokens[m];
  float v = te[(size_t)tok * D_ + d] + pe[(size_t)s * D_ + d];
  x[idx]  = v;
  xh[idx] = (_Float16)v;
}

// ---------------------------------------------------------------------------
// Wt[n][k] = (f16) W[k][n]   (coalesced read, transposed f16 write)
// ---------------------------------------------------------------------------
__global__ __launch_bounds__(256) void convT_kernel(
    const float* __restrict__ W, _Float16* __restrict__ Wt, int Krows, int N)
{
  size_t idx = (size_t)blockIdx.x * 256 + threadIdx.x;   // over Krows*N
  size_t k = idx / (size_t)N;
  size_t n = idx % (size_t)N;
  Wt[n * (size_t)Krows + k] = (_Float16)W[idx];
}

__global__ __launch_bounds__(256) void f2h_kernel(
    const float* __restrict__ in, _Float16* __restrict__ out)
{
  size_t idx = (size_t)blockIdx.x * 256 + threadIdx.x;
  out[idx] = (_Float16)in[idx];
}

// ---------------------------------------------------------------------------
// repack [B*S][D] (d = h*64+hd) -> q/k: [B,H,S,64]   v: [B,H,64,S] (transposed)
// ---------------------------------------------------------------------------
__global__ __launch_bounds__(256) void repack_qk_kernel(
    const _Float16* __restrict__ raw, _Float16* __restrict__ out)
{
  size_t idx = (size_t)blockIdx.x * 256 + threadIdx.x;   // over B*H*S*64
  int hd = (int)(idx & 63);
  int s  = (int)((idx >> 6) & (S_ - 1));
  int h  = (int)((idx >> 16) & (H_ - 1));
  int b  = (int)(idx >> 20);
  out[idx] = raw[((size_t)(b * S_ + s)) * D_ + h * HD_ + hd];
}

__global__ __launch_bounds__(256) void repack_v_kernel(
    const _Float16* __restrict__ raw, _Float16* __restrict__ out)
{
  size_t idx = (size_t)blockIdx.x * 256 + threadIdx.x;   // over B*H*64*S
  int s  = (int)(idx & (S_ - 1));
  int hd = (int)((idx >> 10) & 63);
  int h  = (int)((idx >> 16) & (H_ - 1));
  int b  = (int)(idx >> 20);
  out[idx] = raw[((size_t)(b * S_ + s)) * D_ + h * HD_ + hd];
}

// ---------------------------------------------------------------------------
// WMMA GEMM: C[M][N] = A[M][K](f16) * Bt[N][K](f16)^T (+bias)(+relu)
// 128x128 block tile, BK=32, 8 waves (2Mx4N), 64x32 per wave.
// Async double-buffered global->LDS pipeline on CDNA5.
// ---------------------------------------------------------------------------
template<typename OUT_T, bool HAS_BIAS, bool RELU>
__global__ __launch_bounds__(256) void gemm_kernel(
    const _Float16* __restrict__ A,
    const _Float16* __restrict__ Bt,
    const float* __restrict__ bias,
    OUT_T* __restrict__ C,
    int M, int N, int K)
{
  __shared__ _Float16 sA[2][128][32];
  __shared__ _Float16 sB[2][128][32];
  const int tid  = threadIdx.x;
  const int lane = tid & 31, wave = tid >> 5;
  const int ln = lane & 15, lg = lane >> 4;
  const int bm = blockIdx.x * 128;
  const int bn = blockIdx.y * 128;
  const int wm = (wave & 1) * 64;
  const int wn = (wave >> 1) * 32;
  const int r0 = tid >> 2;           // 0..63, thread also covers r0+64
  const int c0 = (tid & 3) * 8;

  v8f acc[4][2] = {};
  const int nk = K >> 5;

#if ASYNC_LDS
  auto issue_tile = [&](int ib, int k0) {
    async_cp16(&A [(size_t)(bm + r0)      * K + k0 + c0], &sA[ib][r0][c0]);
    async_cp16(&A [(size_t)(bm + r0 + 64) * K + k0 + c0], &sA[ib][r0 + 64][c0]);
    async_cp16(&Bt[(size_t)(bn + r0)      * K + k0 + c0], &sB[ib][r0][c0]);
    async_cp16(&Bt[(size_t)(bn + r0 + 64) * K + k0 + c0], &sB[ib][r0 + 64][c0]);
  };
  issue_tile(0, 0);
  for (int i = 0; i < nk; ++i) {
    async_wait0();                   // my buf[i&1] writes landed in LDS
    __syncthreads();                 // everyone's landed; prev reads of other buf done
    if (i + 1 < nk) issue_tile((i + 1) & 1, (i + 1) << 5);
    const int ib = i & 1;
    Frag af[4], bf[2];
#pragma unroll
    for (int mt = 0; mt < 4; ++mt) {
      int row = wm + mt * 16 + ln;
      af[mt].h[0] = *(const v8h*)&sA[ib][row][lg * 8];
      af[mt].h[1] = *(const v8h*)&sA[ib][row][16 + lg * 8];
    }
#pragma unroll
    for (int nt = 0; nt < 2; ++nt) {
      int row = wn + nt * 16 + ln;
      bf[nt].h[0] = *(const v8h*)&sB[ib][row][lg * 8];
      bf[nt].h[1] = *(const v8h*)&sB[ib][row][16 + lg * 8];
    }
#pragma unroll
    for (int mt = 0; mt < 4; ++mt)
#pragma unroll
      for (int nt = 0; nt < 2; ++nt)
        acc[mt][nt] = wmma_f16(af[mt], bf[nt], acc[mt][nt]);
  }
#else
  for (int i = 0; i < nk; ++i) {
    const int k0 = i << 5;
    *(v8h*)&sA[0][r0][c0]      = *(const v8h*)&A [(size_t)(bm + r0)      * K + k0 + c0];
    *(v8h*)&sA[0][r0 + 64][c0] = *(const v8h*)&A [(size_t)(bm + r0 + 64) * K + k0 + c0];
    *(v8h*)&sB[0][r0][c0]      = *(const v8h*)&Bt[(size_t)(bn + r0)      * K + k0 + c0];
    *(v8h*)&sB[0][r0 + 64][c0] = *(const v8h*)&Bt[(size_t)(bn + r0 + 64) * K + k0 + c0];
    if (k0 + 32 < K) {
      __builtin_prefetch(&A [(size_t)(bm + (tid >> 1)) * K + k0 + 32], 0, 0);
      __builtin_prefetch(&Bt[(size_t)(bn + (tid >> 1)) * K + k0 + 32], 0, 0);
    }
    __syncthreads();
    Frag af[4], bf[2];
#pragma unroll
    for (int mt = 0; mt < 4; ++mt) {
      int row = wm + mt * 16 + ln;
      af[mt].h[0] = *(const v8h*)&sA[0][row][lg * 8];
      af[mt].h[1] = *(const v8h*)&sA[0][row][16 + lg * 8];
    }
#pragma unroll
    for (int nt = 0; nt < 2; ++nt) {
      int row = wn + nt * 16 + ln;
      bf[nt].h[0] = *(const v8h*)&sB[0][row][lg * 8];
      bf[nt].h[1] = *(const v8h*)&sB[0][row][16 + lg * 8];
    }
#pragma unroll
    for (int mt = 0; mt < 4; ++mt)
#pragma unroll
      for (int nt = 0; nt < 2; ++nt)
        acc[mt][nt] = wmma_f16(af[mt], bf[nt], acc[mt][nt]);
    __syncthreads();
  }
#endif

#pragma unroll
  for (int nt = 0; nt < 2; ++nt) {
    const int ncol = bn + wn + nt * 16 + ln;
    const float bv = HAS_BIAS ? bias[ncol] : 0.0f;
#pragma unroll
    for (int mt = 0; mt < 4; ++mt) {
#pragma unroll
      for (int r = 0; r < 8; ++r) {
        const int mrow = bm + wm + mt * 16 + lg * 8 + r;
        float v = acc[mt][nt][r] + bv;
        if (RELU) v = fmaxf(v, 0.0f);
        C[(size_t)mrow * N + ncol] = (OUT_T)v;
      }
    }
  }
}

// ---------------------------------------------------------------------------
// Flash attention, one wave per (b,h,16-query block). Computes S^T = K*Q^T so
// the softmax row lives in the lane dim (single shfl_xor reduction) and
// exp(S^T) is already laid out as the B-fragment of O^T = V^T * P^T.
// K/V fragments are preloaded at iteration top so loads overlap the exp chain.
// ---------------------------------------------------------------------------
__global__ __launch_bounds__(256) void attn_kernel(
    const _Float16* __restrict__ qh,   // [B,H,S,64]
    const _Float16* __restrict__ kh,   // [B,H,S,64]
    const _Float16* __restrict__ vt,   // [B,H,64,S]
    float* __restrict__ o)             // [B,S,D]
{
  const int tid  = threadIdx.x;
  const int lane = tid & 31, ln = lane & 15, lg = lane >> 4;
  const int w    = blockIdx.x * 8 + (tid >> 5);
  const int NQB  = S_ / 16;
  const int qblk = w % NQB;
  const int h    = (w / NQB) % H_;
  const int b    = w / (NQB * H_);
  const int qbase = qblk * 16;
  const int q     = qbase + ln;        // this lane's query row

  const _Float16* qrow = qh + (((size_t)(b * H_ + h)) * S_ + q) * HD_;
  Frag qf[2];                          // Q^T B-fragments, kk = 0 / 32
  qf[0].h[0] = *(const v8h*)(qrow + lg * 8);
  qf[0].h[1] = *(const v8h*)(qrow + 16 + lg * 8);
  qf[1].h[0] = *(const v8h*)(qrow + 32 + lg * 8);
  qf[1].h[1] = *(const v8h*)(qrow + 48 + lg * 8);

  const _Float16* kbasep = kh + ((size_t)(b * H_ + h)) * S_ * HD_;
  const _Float16* vbasep = vt + ((size_t)(b * H_ + h)) * HD_ * S_;

  v8f vacc[4] = {};                    // O^T accumulators (4 hd-tiles)
  float mrun = -3.0e38f, lrun = 0.0f;
  const float scale = 0.125f;          // HD^-0.5
  const int kmax = qbase + 15;

  for (int kb = 0; kb <= kmax; kb += 32) {
    // -------- preload all K and V fragments for this 32-key block ----------
    Frag kf[2][2], vf[4];
#pragma unroll
    for (int kt = 0; kt < 2; ++kt) {
      const _Float16* krow = kbasep + (size_t)(kb + kt * 16 + ln) * HD_;
#pragma unroll
      for (int kk = 0; kk < 2; ++kk) {
        kf[kt][kk].h[0] = *(const v8h*)(krow + kk * 32 + lg * 8);
        kf[kt][kk].h[1] = *(const v8h*)(krow + kk * 32 + 16 + lg * 8);
      }
    }
#pragma unroll
    for (int t = 0; t < 4; ++t) {
      const _Float16* vrow = vbasep + (size_t)(t * 16 + ln) * S_ + kb;
      vf[t].h[0] = *(const v8h*)(vrow + lg * 8);
      vf[t].h[1] = *(const v8h*)(vrow + 16 + lg * 8);
    }
    if (kb + 32 <= kmax) {             // hint next key block into cache
      __builtin_prefetch(kbasep + (size_t)(kb + 32 + ln) * HD_, 0, 0);
      __builtin_prefetch(vbasep + (size_t)ln * S_ + kb + 32, 0, 0);
    }
    // -------- S^T = K * Q^T ------------------------------------------------
    v8f st[2] = {};
#pragma unroll
    for (int kt = 0; kt < 2; ++kt)
#pragma unroll
      for (int kk = 0; kk < 2; ++kk)
        st[kt] = wmma_f16(kf[kt][kk], qf[kk], st[kt]);
    // -------- masked scale + online softmax (per query == per lane) --------
    float mloc = -3.0e38f;
    float sv[2][8];
#pragma unroll
    for (int kt = 0; kt < 2; ++kt)
#pragma unroll
      for (int r = 0; r < 8; ++r) {
        int key = kb + kt * 16 + lg * 8 + r;
        float s = (key <= q) ? st[kt][r] * scale : -3.0e38f;
        sv[kt][r] = s;
        mloc = fmaxf(mloc, s);
      }
    mloc = fmaxf(mloc, __shfl_xor(mloc, 16, 32));
    float mnew  = fmaxf(mrun, mloc);
    float alpha = __expf(mrun - mnew);
    Frag pb;                           // P^T B-fragment: no lane movement!
    float ps = 0.0f;
#pragma unroll
    for (int kt = 0; kt < 2; ++kt)
#pragma unroll
      for (int r = 0; r < 8; ++r) {
        float e = __expf(sv[kt][r] - mnew);
        ps += e;
        pb.v[kt * 8 + r] = (_Float16)e;
      }
    ps += __shfl_xor(ps, 16, 32);
    lrun = lrun * alpha + ps;
    mrun = mnew;
    // -------- O^T += V^T * P^T --------------------------------------------
#pragma unroll
    for (int t = 0; t < 4; ++t) {
#pragma unroll
      for (int r = 0; r < 8; ++r) vacc[t][r] *= alpha;
      vacc[t] = wmma_f16(vf[t], pb, vacc[t]);
    }
  }
  const float invl = 1.0f / lrun;
  float* orow = o + ((size_t)(b * S_ + q)) * D_ + h * HD_;
#pragma unroll
  for (int t = 0; t < 4; ++t)
#pragma unroll
    for (int r = 0; r < 8; ++r)
      orow[t * 16 + lg * 8 + r] = vacc[t][r] * invl;
}

// ---------------------------------------------------------------------------
// out = (ADDX ? x : 0) + LayerNorm(y)*gamma+beta ; one block per row (D=1024)
// ---------------------------------------------------------------------------
template<bool ADDX, bool WF32, bool WF16>
__global__ __launch_bounds__(256) void addln_kernel(
    const float* __restrict__ xres,
    const float* __restrict__ y,
    const float* __restrict__ gam,
    const float* __restrict__ bet,
    float* __restrict__ o32,
    _Float16* __restrict__ o16)
{
  const int row = blockIdx.x;
  const int tid = threadIdx.x;
  const int lane = tid & 31, wave = tid >> 5;
  const float* yr = y + (size_t)row * D_;
  float v0[4]; float sum = 0.0f, sq = 0.0f;
#pragma unroll
  for (int i = 0; i < 4; ++i) {
    float v = yr[tid + i * 256];
    v0[i] = v; sum += v; sq += v * v;
  }
#pragma unroll
  for (int off = 16; off > 0; off >>= 1) {
    sum += __shfl_xor(sum, off, 32);
    sq  += __shfl_xor(sq,  off, 32);
  }
  __shared__ float s1[8], s2[8];
  if (lane == 0) { s1[wave] = sum; s2[wave] = sq; }
  __syncthreads();
  sum = 0.0f; sq = 0.0f;
#pragma unroll
  for (int wv = 0; wv < 8; ++wv) { sum += s1[wv]; sq += s2[wv]; }
  const float mean = sum * (1.0f / D_);
  const float var  = sq * (1.0f / D_) - mean * mean;
  const float rstd = rsqrtf(var + 1e-5f);
#pragma unroll
  for (int i = 0; i < 4; ++i) {
    int d = tid + i * 256;
    float v = (v0[i] - mean) * rstd * gam[d] + bet[d];
    if (ADDX) v += xres[(size_t)row * D_ + d];
    if (WF32) o32[(size_t)row * D_ + d] = v;
    if (WF16) o16[(size_t)row * D_ + d] = (_Float16)v;
  }
}

// ---------------------------------------------------------------------------
extern "C" void kernel_launch(void* const* d_in, const int* in_sizes, int n_in,
                              void* d_out, int out_size, void* d_ws, size_t ws_size,
                              hipStream_t stream) {
  (void)in_sizes; (void)n_in; (void)out_size; (void)ws_size;
  const int*   tokens = (const int*)d_in[0];
  const float* tok_emb= (const float*)d_in[1];
  const float* pos_emb= (const float*)d_in[2];
  const float* Wq  = (const float*)d_in[3];
  const float* Wk  = (const float*)d_in[4];
  const float* Wv  = (const float*)d_in[5];
  const float* Wp  = (const float*)d_in[6];
  const float* ln1s= (const float*)d_in[7];
  const float* ln1b= (const float*)d_in[8];
  const float* W1  = (const float*)d_in[9];
  const float* b1  = (const float*)d_in[10];
  const float* W2  = (const float*)d_in[11];
  const float* b2  = (const float*)d_in[12];
  const float* ln2s= (const float*)d_in[13];
  const float* ln2b= (const float*)d_in[14];
  const float* lnfs= (const float*)d_in[15];
  const float* lnfb= (const float*)d_in[16];
  const float* Wh  = (const float*)d_in[17];
  const float* bh  = (const float*)d_in[18];

  char* ws = (char*)d_ws;
  const size_t MB = 1ull << 20;
  float*    x    = (float*)   (ws +   0 * MB);   // 16MB  [M,D] f32 (persistent)
  _Float16* xh   = (_Float16*)(ws +  16 * MB);   //  8MB  [M,D] f16
  float*    res  = (float*)   (ws +  24 * MB);   // 16MB  GEMM f32 out (proj/ffn2)
  _Float16* ah   = (_Float16*)(ws +  40 * MB);   //  8MB  ln1 out f16
  _Float16* qh   = (_Float16*)(ws +  48 * MB);   //  8MB  [B,H,S,64]
  _Float16* kh   = (_Float16*)(ws +  56 * MB);   //  8MB
  _Float16* vth  = (_Float16*)(ws +  64 * MB);   //  8MB  [B,H,64,S]
  _Float16* qraw = (_Float16*)(ws +  72 * MB);   //  8MB
  _Float16* kraw = (_Float16*)(ws +  80 * MB);   //  8MB
  _Float16* vraw = (_Float16*)(ws +  88 * MB);   //  8MB
  float*    of   = (float*)   (ws +  96 * MB);   // 16MB  attention out f32
  _Float16* oh   = (_Float16*)(ws + 112 * MB);   //  8MB
  _Float16* hh   = (_Float16*)(ws + 120 * MB);   // 32MB  ffn hidden f16
  _Float16* wt   = (_Float16*)(ws + 152 * MB);   // 64MB  current weight f16^T

  const dim3 blk(256);
  const int  gEmb  = M_ * D_ / 256;
  const int  gRep  = B_ * H_ * S_ * HD_ / 256;
  const int  gAttn = B_ * H_ * (S_ / 16) / 8;
  const dim3 gDD(M_ / 128, D_ / 128);
  const dim3 gDF(M_ / 128, F_ / 128);
  const dim3 gDV(M_ / 128, V_ / 128);

  embed_kernel<<<gEmb, blk, 0, stream>>>(tokens, tok_emb, pos_emb, x, xh);

  for (int l = 0; l < L_; ++l) {
    const size_t oDD = (size_t)l * D_ * D_;
    // ---- QKV projections ----
    convT_kernel<<<D_ * D_ / 256, blk, 0, stream>>>(Wq + oDD, wt, D_, D_);
    gemm_kernel<_Float16,false,false><<<gDD, blk, 0, stream>>>(xh, wt, nullptr, qraw, M_, D_, D_);
    convT_kernel<<<D_ * D_ / 256, blk, 0, stream>>>(Wk + oDD, wt, D_, D_);
    gemm_kernel<_Float16,false,false><<<gDD, blk, 0, stream>>>(xh, wt, nullptr, kraw, M_, D_, D_);
    convT_kernel<<<D_ * D_ / 256, blk, 0, stream>>>(Wv + oDD, wt, D_, D_);
    gemm_kernel<_Float16,false,false><<<gDD, blk, 0, stream>>>(xh, wt, nullptr, vraw, M_, D_, D_);
    repack_qk_kernel<<<gRep, blk, 0, stream>>>(qraw, qh);
    repack_qk_kernel<<<gRep, blk, 0, stream>>>(kraw, kh);
    repack_v_kernel <<<gRep, blk, 0, stream>>>(vraw, vth);
    // ---- attention ----
    attn_kernel<<<gAttn, blk, 0, stream>>>(qh, kh, vth, of);
    f2h_kernel<<<M_ * D_ / 256, blk, 0, stream>>>(of, oh);
    // ---- output projection + residual-LN1 ----
    convT_kernel<<<D_ * D_ / 256, blk, 0, stream>>>(Wp + oDD, wt, D_, D_);
    gemm_kernel<float,false,false><<<gDD, blk, 0, stream>>>(oh, wt, nullptr, res, M_, D_, D_);
    addln_kernel<true,false,true><<<M_, blk, 0, stream>>>(x, res, ln1s + (size_t)l * D_,
                                                          ln1b + (size_t)l * D_, nullptr, ah);
    // ---- FFN ----
    convT_kernel<<<D_ * F_ / 256, blk, 0, stream>>>(W1 + (size_t)l * D_ * F_, wt, D_, F_);
    gemm_kernel<_Float16,true,true><<<gDF, blk, 0, stream>>>(ah, wt, b1 + (size_t)l * F_, hh, M_, F_, D_);
    convT_kernel<<<F_ * D_ / 256, blk, 0, stream>>>(W2 + (size_t)l * F_ * D_, wt, F_, D_);
    gemm_kernel<float,true,false><<<gDD, blk, 0, stream>>>(hh, wt, b2 + (size_t)l * D_, res, M_, D_, F_);
    // ---- residual-LN2 (residual from block input, in-place x update) ----
    addln_kernel<true,true,true><<<M_, blk, 0, stream>>>(x, res, ln2s + (size_t)l * D_,
                                                         ln2b + (size_t)l * D_, x, xh);
  }

  // ---- final LN + LM head ----
  addln_kernel<false,false,true><<<M_, blk, 0, stream>>>(nullptr, x, lnfs, lnfb, nullptr, xh);
  convT_kernel<<<(int)((size_t)D_ * V_ / 256), blk, 0, stream>>>(Wh, wt, D_, V_);
  gemm_kernel<float,true,false><<<gDV, blk, 0, stream>>>(xh, wt, bh, (float*)d_out, M_, V_, D_);
}
